// WMA_2662879723836
// MI455X (gfx1250) — matrix operand order repaired and checked by reading, not compiled, not run on hardware
//
#include <hip/hip_runtime.h>
#include <hip/hip_bf16.h>

typedef __attribute__((ext_vector_type(16))) __bf16 v16bf;
typedef __attribute__((ext_vector_type(8)))  float  v8f;
typedef __attribute__((ext_vector_type(4)))  unsigned int u32x4;
typedef __attribute__((ext_vector_type(8)))  int i32x8;
typedef __attribute__((ext_vector_type(4)))  int i32x4;

#define B_   2
#define DIM  128
#define RD   32
#define HF   256
#define WF   256
#define HH   128
#define WH   128
#define NP   (HH*WH)        // 16384 half-res positions
#define HEADS 8
#define GRP  16
#define GC   8

// CDNA5 wave32 WMMA fragment index maps (CDNA5 ISA §7.12.2)
// A (16x32 bf16): lane m = L&15; elem i<8 -> K = (L>>4)*8+i ; i>=8 -> K = 16+(L>>4)*8+(i-8)
__device__ __forceinline__ int kA_idx(int i, int kh) {
  return (i < 8) ? (kh * 8 + i) : (16 + kh * 8 + (i - 8));
}
// B (32x16 bf16): lane n = L&15; elem i<8 -> K = (L>>4)*8+i ; i>=8 -> K = 16+(L>>4)*8+(i-8)
__device__ __forceinline__ int kB_idx(int i, int kh) { return (i < 8) ? (kh * 8 + i) : (16 + kh * 8 + (i - 8)); }

__device__ __forceinline__ v8f wmma_bf16(v16bf a, v16bf b, v8f c) {
  return __builtin_amdgcn_wmma_f32_16x16x32_bf16(false, a, false, b, (short)0, c,
                                                 false, false);
}

// TDM tile load (up to 3-D): tile_d0 contiguous elements, tile_d1 lines at
// stride0 elements, tile_d2 planes at stride1 elements. Descriptor packing per
// CDNA5 ISA. Caller: single fully-active wave,
// then s_wait_tensorcnt + block barrier. 6-arg builtin (clang-23 form).
__device__ __forceinline__ void tdm_load(unsigned lds_off,
                                         unsigned long long gaddr,
                                         unsigned data_size_code,  // 0=1B,1=2B,2=4B
                                         unsigned td0, unsigned td1, unsigned td2,
                                         unsigned t0, unsigned t1, unsigned t2,
                                         unsigned stride0, unsigned stride1) {
  u32x4 g0 = { 1u,                                   // count=1, user descriptor
               lds_off,                               // lds_addr bytes
               (unsigned)(gaddr & 0xFFFFFFFFu),
               (unsigned)(((gaddr >> 32) & 0x01FFFFFFu) | 0x80000000u) }; // type=2
  i32x8 g1;
  g1[0] = (int)(data_size_code << 16);               // wg_mask=0, data_size
  g1[1] = (int)((td0 & 0xFFFFu) << 16);              // tensor_dim0[15:0] @63:48
  g1[2] = (int)(((td0 >> 16) & 0xFFFFu) | ((td1 & 0xFFFFu) << 16));
  g1[3] = (int)(((td1 >> 16) & 0xFFFFu) | ((t0 & 0xFFFFu) << 16));
  g1[4] = (int)((t1 & 0xFFFFu) | ((t2 & 0xFFFFu) << 16));
  g1[5] = (int)stride0;                              // tensor_dim0_stride[31:0]
  g1[6] = (int)((stride1 & 0xFFFFu) << 16);          // s0 hi=0 | s1[15:0]
  g1[7] = (int)(stride1 >> 16);                      // s1[47:16]
  i32x4 g2 = { (int)td2, 0, 0, 0 };                  // tensor_dim2; rest unused
  i32x4 g3 = { 0, 0, 0, 0 };
  i32x8 gz8 = { 0, 0, 0, 0, 0, 0, 0, 0 };
  __builtin_amdgcn_tensor_load_to_lds(g0, g1, g2, g3, gz8, 0);
}

__device__ __forceinline__ unsigned lds_byte_off(const void* p) {
  // generic-AS pointer to LDS: low 32 bits are the LDS byte address
  return (unsigned)(unsigned long long)p;
}

// ---------------------------------------------------------------------------
// K1: fused reduce 1x1 (128->32) + ReLU + Haar DWT.
// Block = 8x8 full-res region: wave0 TDM-loads the 8x8x128ch f32 tile (32 KB)
// into LDS; each of the 8 waves runs a 16-pixel x 32-out-ch GEMM (8 WMMAs)
// with A-fragments from LDS, ReLUs into LDS, Haar-combines, stores bf16.
// ---------------------------------------------------------------------------
__global__ __launch_bounds__(256) void k_reduce_dwt(
    const float* __restrict__ x, const float* __restrict__ rw,
    const float* __restrict__ rb, __bf16* __restrict__ xdwt) {
  __shared__ float stage[DIM][8][8];   // [channel][y][x] f32, 32 KB
  __shared__ float lr[8][16][32];
  const int tid = threadIdx.x;
  const int b = blockIdx.x >> 10;      // 32x32 = 1024 regions per batch
  const int r = blockIdx.x & 1023;
  const int ry = r >> 5, rx = r & 31;
  const int Y0 = ry * 8, X0 = rx * 8;

  if (tid < 32) {
    const float* src = x + (size_t)b * DIM * HF * WF + (size_t)Y0 * WF + X0;
    tdm_load(lds_byte_off(&stage[0][0][0]), (unsigned long long)src,
             /*ds=*/2u, /*td0=*/WF, /*td1=*/HF, /*td2=*/DIM,
             /*t0=*/8u, /*t1=*/8u, /*t2=*/DIM,
             /*s0=*/WF, /*s1=*/HF * WF);
    __builtin_amdgcn_s_wait_tensorcnt(0);
  }
  __syncthreads();

  const int lane = tid & 31, wl = tid >> 5;
  const int wx = wl & 1, wy = wl >> 1;     // 2x4 wave grid of 4x4 tiles
  const int ml = lane & 15, kh = lane >> 4;
  const int py = ml >> 2, px = ml & 3;
  const int ly = wy * 4 + py, lx = wx * 4 + px;

  v16bf a[4];
#pragma unroll
  for (int kc = 0; kc < 4; ++kc)
#pragma unroll
    for (int i = 0; i < 16; ++i)
      a[kc][i] = (__bf16)stage[kc * 32 + kA_idx(i, kh)][ly][lx];

#pragma unroll
  for (int nt = 0; nt < 2; ++nt) {
    const int n = nt * 16 + ml;
    v8f acc = {};
#pragma unroll
    for (int kc = 0; kc < 4; ++kc) {
      v16bf bfr;
#pragma unroll
      for (int i = 0; i < 16; ++i)
        bfr[i] = (__bf16)rw[n * DIM + kc * 32 + kB_idx(i, kh)];
      acc = wmma_bf16(a[kc], bfr, acc);
    }
    const float bias = rb[n];
#pragma unroll
    for (int j = 0; j < 8; ++j) lr[wl][j + kh * 8][n] = fmaxf(acc[j] + bias, 0.f);
  }
  __syncthreads();
  const int ch = lane;  // 32 reduced channels
#pragma unroll
  for (int q = 0; q < 4; ++q) {
    const int qy = q >> 1, qx = q & 1;
    const float r00 = lr[wl][(2 * qy + 0) * 4 + 2 * qx + 0][ch];
    const float r01 = lr[wl][(2 * qy + 0) * 4 + 2 * qx + 1][ch];
    const float r10 = lr[wl][(2 * qy + 1) * 4 + 2 * qx + 0][ch];
    const float r11 = lr[wl][(2 * qy + 1) * 4 + 2 * qx + 1][ch];
    const float ll = 0.5f * (r00 + r01 + r10 + r11);
    const float lh = 0.5f * (r00 + r01 - r10 - r11);
    const float hl = 0.5f * (r00 - r01 + r10 - r11);
    const float hh = 0.5f * (r00 - r01 - r10 + r11);
    const int hy = (Y0 >> 1) + wy * 2 + qy;
    const int hx = (X0 >> 1) + wx * 2 + qx;
    const size_t base = (size_t)b * DIM * NP + (size_t)hy * WH + hx;
    xdwt[base + (size_t)(0 * 32 + ch) * NP] = (__bf16)ll;
    xdwt[base + (size_t)(1 * 32 + ch) * NP] = (__bf16)lh;
    xdwt[base + (size_t)(2 * 32 + ch) * NP] = (__bf16)hl;
    xdwt[base + (size_t)(3 * 32 + ch) * NP] = (__bf16)hh;
  }
}

// ---------------------------------------------------------------------------
// K2: qkv 1x1 (128->384) + fil branch t = relu(W1 @ x_dwt + b1), sharing one
// TDM-staged activation tile (128ch x 16pos bf16). Each of 8 waves does 3 of
// the 24 qkv M-tiles; waves 0/1 additionally emit the 32-channel t GEMM.
// ---------------------------------------------------------------------------
__global__ __launch_bounds__(256) void k_qkv_tgen(
    const __bf16* __restrict__ xdwt, const float* __restrict__ qw,
    const float* __restrict__ w1, const float* __restrict__ b1,
    float* __restrict__ qkv, __bf16* __restrict__ tbf) {
  __shared__ __bf16 stage[DIM][16];
  const int tid = threadIdx.x;
  const int b = blockIdx.x >> 10;      // 1024 pos tiles per batch
  const int pt = blockIdx.x & 1023;
  const int pos = pt * 16;

  if (tid < 32) {  // wave 0 issues the DMA and waits on its TENSORcnt
    const __bf16* src = xdwt + (size_t)b * DIM * NP + pos;
    tdm_load(lds_byte_off(&stage[0][0]), (unsigned long long)src,
             /*ds=*/1u, /*td0=*/NP, /*td1=*/DIM, /*td2=*/1u,
             /*t0=*/16u, /*t1=*/DIM, /*t2=*/0u,
             /*s0=*/NP, /*s1=*/0u);
    __builtin_amdgcn_s_wait_tensorcnt(0);
  }
  __syncthreads();

  const int lane = tid & 31, wl = tid >> 5;
  const int ml = lane & 15, kh = lane >> 4;

  v16bf bfr[4];
#pragma unroll
  for (int kc = 0; kc < 4; ++kc)
#pragma unroll
    for (int i = 0; i < 16; ++i)
      bfr[kc][i] = stage[kc * 32 + kB_idx(i, kh)][ml];

#pragma unroll
  for (int t = 0; t < 3; ++t) {
    const int mt = wl * 3 + t;
    v8f acc = {};
#pragma unroll
    for (int kc = 0; kc < 4; ++kc) {
      v16bf a;
#pragma unroll
      for (int i = 0; i < 16; ++i)
        a[i] = (__bf16)qw[(mt * 16 + ml) * DIM + kc * 32 + kA_idx(i, kh)];
      acc = wmma_bf16(a, bfr[kc], acc);
    }
#pragma unroll
    for (int j = 0; j < 8; ++j) {
      const int m = mt * 16 + j + kh * 8;
      qkv[(size_t)(b * 384 + m) * NP + pos + ml] = acc[j];
    }
  }

  if (wl < 2) {  // fil branch: M=32 -> 2 M-tiles on waves 0/1 (full-wave EXEC)
    v8f acc = {};
#pragma unroll
    for (int kc = 0; kc < 4; ++kc) {
      v16bf a;
#pragma unroll
      for (int i = 0; i < 16; ++i)
        a[i] = (__bf16)w1[(wl * 16 + ml) * DIM + kc * 32 + kA_idx(i, kh)];
      acc = wmma_bf16(a, bfr[kc], acc);
    }
#pragma unroll
    for (int j = 0; j < 8; ++j) {
      const int m = wl * 16 + j + kh * 8;
      tbf[(size_t)(b * RD + m) * NP + pos + ml] = (__bf16)fmaxf(acc[j] + b1[m], 0.f);
    }
  }
}

// ---------------------------------------------------------------------------
// K3: depthwise 3x3 over 384 channels at half resolution (VALU, bandwidth op)
// ---------------------------------------------------------------------------
__global__ __launch_bounds__(256) void k_dw3(const float* __restrict__ in,
                                             const float* __restrict__ w,
                                             float* __restrict__ out) {
  const size_t idx = (size_t)blockIdx.x * 256 + threadIdx.x;
  const int pos = (int)(idx % NP);
  const int rem = (int)(idx / NP);
  const int ch = rem % 384, b = rem / 384;
  const int y = pos >> 7, xq = pos & 127;
  float w9[9];
#pragma unroll
  for (int i = 0; i < 9; ++i) w9[i] = w[ch * 9 + i];
  const float* p = in + (size_t)(b * 384 + ch) * NP;
  float acc = 0.f;
#pragma unroll
  for (int dy = 0; dy < 3; ++dy) {
    const int yy = y + dy - 1;
    if (yy < 0 || yy >= HH) continue;
#pragma unroll
    for (int dx = 0; dx < 3; ++dx) {
      const int xx = xq + dx - 1;
      if (xx < 0 || xx >= WH) continue;
      acc += w9[dy * 3 + dx] * p[yy * WH + xx];
    }
  }
  out[idx] = acc;
}

// ---------------------------------------------------------------------------
// K4a: attention partials. 8 segment-blocks per (b,head) (128 blocks total):
// each block contracts a K-range of 2048 (8 waves x 8 accumulating WMMAs)
// with the row sums-of-squares fused into the fragment loads; the partial
// 16x16 matrix + 32 partial norms go to workspace.
// ---------------------------------------------------------------------------
__global__ __launch_bounds__(256) void k_attn_part(const float* __restrict__ qkv2,
                                                   float* __restrict__ attp) {
  __shared__ float part[8][16][16];
  __shared__ float sqq[8][32];
  __shared__ float sqk[8][32];
  const int tid = threadIdx.x;
  const int bh = blockIdx.x >> 3, seg = blockIdx.x & 7;
  const int b = bh >> 3, head = bh & 7;
  const float* q = qkv2 + (size_t)(b * 384 + head * 16) * NP;
  const float* kmat = qkv2 + (size_t)(b * 384 + 128 + head * 16) * NP;

  const int lane = tid & 31, wl = tid >> 5;
  const int ml = lane & 15, kh = lane >> 4;
  const int kseg = seg * 2048 + wl * 256;
  float sq = 0.f, sk = 0.f;
  v8f acc = {};
#pragma unroll
  for (int kc = 0; kc < 8; ++kc) {
    const int kbase = kseg + kc * 32;
    v16bf a, bfr;
#pragma unroll
    for (int i = 0; i < 16; ++i) {
      const float xv = q[(size_t)ml * NP + kbase + kA_idx(i, kh)];
      sq += xv * xv;
      a[i] = (__bf16)xv;
    }
#pragma unroll
    for (int i = 0; i < 16; ++i) {
      const float xv = kmat[(size_t)ml * NP + kbase + kB_idx(i, kh)];
      sk += xv * xv;
      bfr[i] = (__bf16)xv;
    }
    acc = wmma_bf16(a, bfr, acc);
  }
#pragma unroll
  for (int j = 0; j < 8; ++j) part[wl][j + kh * 8][ml] = acc[j];
  sqq[wl][lane] = sq;
  sqk[wl][lane] = sk;
  __syncthreads();
  float* dst = attp + (size_t)blockIdx.x * 288;
  {
    const int m = tid >> 4, n = tid & 15;
    float s = 0.f;
#pragma unroll
    for (int w8 = 0; w8 < 8; ++w8) s += part[w8][m][n];
    dst[m * 16 + n] = s;
  }
  if (tid < 32) {
    // rows 0..15 = q rows, 16..31 = k rows; lanes (r, r+16) tile K exactly
    float s = 0.f;
    const int r = tid & 15;
#pragma unroll
    for (int w8 = 0; w8 < 8; ++w8)
      s += (tid < 16) ? (sqq[w8][r] + sqq[w8][r + 16])
                      : (sqk[w8][r] + sqk[w8][r + 16]);
    dst[256 + tid] = s;
  }
}

// ---------------------------------------------------------------------------
// K4b: reduce the 8 segments per (b,head), apply 1/||q|| 1/||k|| *
// temperature, ReLU -> final 16x16 attention matrices.
// ---------------------------------------------------------------------------
__global__ __launch_bounds__(256) void k_attn_fin(const float* __restrict__ attp,
                                                  const float* __restrict__ temp,
                                                  float* __restrict__ attn_ws) {
  __shared__ float invn[32];
  const int tid = threadIdx.x;
  const int bh = blockIdx.x;
  const int head = bh & 7;
  if (tid < 32) {
    float s = 0.f;
#pragma unroll
    for (int seg = 0; seg < 8; ++seg)
      s += attp[(size_t)(bh * 8 + seg) * 288 + 256 + tid];
    invn[tid] = 1.f / fmaxf(sqrtf(s), 1e-12f);
  }
  __syncthreads();
  const int m = tid >> 4, n = tid & 15;
  float s = 0.f;
#pragma unroll
  for (int seg = 0; seg < 8; ++seg)
    s += attp[(size_t)(bh * 8 + seg) * 288 + m * 16 + n];
  s *= temp[head] * invn[m] * invn[16 + n];
  attn_ws[bh * 256 + m * 16 + n] = fmaxf(s, 0.f);
}

// ---------------------------------------------------------------------------
// K4c: out = attn @ v, spread over the whole GPU (1024 blocks). Each block
// serves 256 positions of one (b,head); attn 16x16 cached in LDS.
// ---------------------------------------------------------------------------
__global__ __launch_bounds__(256) void k_attn_apply(const float* __restrict__ qkv2,
                                                    const float* __restrict__ attn_ws,
                                                    float* __restrict__ ao) {
  __shared__ float attn_s[16][16];
  const int tid = threadIdx.x;
  const int bh = blockIdx.x >> 6;      // 0..15
  const int chunk = blockIdx.x & 63;   // 64 chunks of 256 positions
  const int b = bh >> 3, head = bh & 7;
  attn_s[tid >> 4][tid & 15] = attn_ws[bh * 256 + tid];
  __syncthreads();
  const float* v = qkv2 + (size_t)(b * 384 + 256 + head * 16) * NP;
  const int pos = chunk * 256 + tid;
  float vv[16];
#pragma unroll
  for (int d = 0; d < 16; ++d) vv[d] = v[(size_t)d * NP + pos];
#pragma unroll
  for (int m = 0; m < 16; ++m) {
    float s = 0.f;
#pragma unroll
    for (int d = 0; d < 16; ++d) s += attn_s[m][d] * vv[d];
    ao[(size_t)(b * DIM + head * 16 + m) * NP + pos] = s;
  }
}

// ---------------------------------------------------------------------------
// K5: dynamic grouped 7x7 conv. Block = (b, group, 8x8 pixel tile): halo is
// 8ch x 14x14 (3.1x re-read vs 6.25x for 4x4 tiles); waves 0..3 each build
// the 49x16 wgen slab for their 4x4 subtile with 4 WMMAs (w2_g[49x32] @
// t[32x16]); 512 outputs applied 2 per thread. wgen never hits HBM.
// ---------------------------------------------------------------------------
__global__ __launch_bounds__(256) void k_dynconv(
    const float* __restrict__ ao, const __bf16* __restrict__ tbf,
    const float* __restrict__ w2, const float* __restrict__ b2,
    float* __restrict__ dyn) {
  __shared__ float halo[GC][14][14];
  __shared__ float wg[64][64];         // [tap row][subtile*16 + n]
  const int tid = threadIdx.x;
  int blk = blockIdx.x;
  const int tile = blk & 255; blk >>= 8;   // 16x16 tiles of 8x8
  const int g = blk & 15;      blk >>= 4;
  const int b = blk;
  const int ty = tile >> 4, tx = tile & 15;
  const int y0 = ty * 8, x0 = tx * 8;

  for (int i = tid; i < GC * 196; i += 256) {
    const int c = i / 196, rr = (i % 196) / 14, cc = i % 14;
    const int y = y0 - 3 + rr, xx = x0 - 3 + cc;
    float val = 0.f;
    if (y >= 0 && y < HH && xx >= 0 && xx < WH)
      val = ao[(size_t)(b * DIM + g * GC + c) * NP + y * WH + xx];
    halo[c][rr][cc] = val;
  }
  const int lane = tid & 31, wl = tid >> 5;
  if (wl < 4) {  // waves 0..3: wgen for 4x4 subtile (sy,sx); full-wave EXEC
    const int sy = wl >> 1, sx = wl & 1;
    const int ml = lane & 15, kh = lane >> 4;
    const int qy = ml >> 2, qx = ml & 3;
    const int pos = (y0 + sy * 4 + qy) * WH + (x0 + sx * 4 + qx);
    v16bf bfr;
#pragma unroll
    for (int i = 0; i < 16; ++i)
      bfr[i] = tbf[(size_t)(b * RD + kB_idx(i, kh)) * NP + pos];
#pragma unroll
    for (int mt = 0; mt < 4; ++mt) {
      v16bf a = {};
      const int row = mt * 16 + ml;
      if (row < 49) {
#pragma unroll
        for (int i = 0; i < 16; ++i)
          a[i] = (__bf16)w2[(g * 49 + row) * RD + kA_idx(i, kh)];
      }
      v8f acc = {};
      acc = wmma_bf16(a, bfr, acc);
#pragma unroll
      for (int j = 0; j < 8; ++j) {
        const int rw_ = mt * 16 + j + kh * 8;
        const float bias = (rw_ < 49) ? b2[g * 49 + rw_] : 0.f;
        wg[rw_][wl * 16 + ml] = acc[j] + bias;
      }
    }
  }
  __syncthreads();
  for (int o = tid; o < GC * 64; o += 256) {   // 512 outputs, 2 per thread
    const int c = o >> 6, p = o & 63;
    const int py = p >> 3, px = p & 7;
    const int col = ((py >> 2) * 2 + (px >> 2)) * 16 + (py & 3) * 4 + (px & 3);
    float acc = 0.f;
#pragma unroll
    for (int di = 0; di < 7; ++di)
#pragma unroll
      for (int dj = 0; dj < 7; ++dj)
        acc += halo[c][py + di][px + dj] * wg[di * 7 + dj][col];
    dyn[(size_t)(b * DIM + g * GC + c) * NP + (y0 + py) * WH + (x0 + px)] = acc;
  }
}

// ---------------------------------------------------------------------------
// K6: fused inverse Haar DWT + proj 1x1 (32->128). Block = 16x8 full-res
// region; wave0 TDM-loads the matching 8x4x128ch f32 half-res tile (16 KB,
// all four subbands = channel blocks at stride NP). Each wave Haar-recombines
// one 16-pixel strip from LDS into a B fragment, then 8 WMMAs over M=128.
// ---------------------------------------------------------------------------
__global__ __launch_bounds__(256) void k_idwt_proj(const float* __restrict__ dyn,
                                                   const float* __restrict__ pw,
                                                   float* __restrict__ out) {
  __shared__ float stage[DIM][4][8];   // [channel][hy][hx] f32, 16 KB
  const int tid = threadIdx.x;
  const int b = blockIdx.x >> 9;       // 32x16 = 512 regions per batch
  const int r = blockIdx.x & 511;
  const int ry = r >> 4, rx = r & 15;
  const int Y0 = ry * 8, X0 = rx * 16;

  if (tid < 32) {
    const float* src = dyn + (size_t)b * DIM * NP + (size_t)(Y0 >> 1) * WH + (X0 >> 1);
    tdm_load(lds_byte_off(&stage[0][0][0]), (unsigned long long)src,
             /*ds=*/2u, /*td0=*/WH, /*td1=*/HH, /*td2=*/DIM,
             /*t0=*/8u, /*t1=*/4u, /*t2=*/DIM,
             /*s0=*/WH, /*s1=*/NP);
    __builtin_amdgcn_s_wait_tensorcnt(0);
  }
  __syncthreads();

  const int lane = tid & 31, wl = tid >> 5;
  const int ml = lane & 15, kh = lane >> 4;
  const int Y = Y0 + wl;               // each wave: one 16-pixel row strip
  const int X = X0 + ml;
  const int d = Y & 1, e = X & 1;
  const int hyl = wl >> 1, hxl = ml >> 1;
  const float shs = d ? -1.f : 1.f;
  const float ses = e ? -1.f : 1.f;
  v16bf bfr;
#pragma unroll
  for (int i = 0; i < 16; ++i) {
    const int c = kB_idx(i, kh);
    const float ll = stage[0 * 32 + c][hyl][hxl];
    const float lh = stage[1 * 32 + c][hyl][hxl];
    const float hl = stage[2 * 32 + c][hyl][hxl];
    const float hh = stage[3 * 32 + c][hyl][hxl];
    bfr[i] = (__bf16)(0.5f * (ll + shs * lh + ses * hl + shs * ses * hh));
  }
#pragma unroll
  for (int mt = 0; mt < 8; ++mt) {
    v16bf a;
#pragma unroll
    for (int i = 0; i < 16; ++i)
      a[i] = (__bf16)pw[(mt * 16 + ml) * RD + kA_idx(i, kh)];
    v8f acc = {};
    acc = wmma_bf16(a, bfr, acc);
#pragma unroll
    for (int j = 0; j < 8; ++j) {
      const int m = mt * 16 + j + kh * 8;
      out[(size_t)(b * DIM + m) * ((size_t)HF * WF) + (size_t)Y * WF + X] = acc[j];
    }
  }
}

// ---------------------------------------------------------------------------
extern "C" void kernel_launch(void* const* d_in, const int* in_sizes, int n_in,
                              void* d_out, int out_size, void* d_ws, size_t ws_size,
                              hipStream_t stream) {
  (void)in_sizes; (void)n_in; (void)out_size; (void)ws_size;
  const float* x    = (const float*)d_in[0];
  const float* rw   = (const float*)d_in[1];
  const float* rb   = (const float*)d_in[2];
  const float* qw   = (const float*)d_in[3];
  const float* qdw  = (const float*)d_in[4];
  const float* temp = (const float*)d_in[5];
  const float* w1   = (const float*)d_in[6];
  const float* b1   = (const float*)d_in[7];
  const float* w2   = (const float*)d_in[8];
  const float* b2   = (const float*)d_in[9];
  const float* pw   = (const float*)d_in[10];
  float* out = (float*)d_out;

  char* ws = (char*)d_ws;
  size_t off = 0;
  auto take = [&](size_t bytes) -> char* {
    char* p = ws + off;
    off = (off + bytes + 255) & ~(size_t)255;
    return p;
  };
  __bf16* xdwt = (__bf16*)take((size_t)B_ * DIM * NP * 2);  // 8.4 MB
  float*  qkv  = (float*) take((size_t)B_ * 384 * NP * 4);  // 50 MB
  float*  qkv2 = (float*) take((size_t)B_ * 384 * NP * 4);  // 50 MB
  float*  aob  = (float*) take((size_t)B_ * DIM * NP * 4);  // 16.8 MB
  __bf16* tbf  = (__bf16*)take((size_t)B_ * RD * NP * 2);   // 2 MB
  float*  dyn  = (float*) take((size_t)B_ * DIM * NP * 4);  // 16.8 MB
  float*  attp = (float*) take((size_t)B_ * HEADS * 8 * 288 * 4);
  float*  attw = (float*) take((size_t)B_ * HEADS * 256 * 4);

  k_reduce_dwt<<<B_ * 1024, 256, 0, stream>>>(x, rw, rb, xdwt);
  k_qkv_tgen<<<B_ * 1024, 256, 0, stream>>>(xdwt, qw, w1, b1, qkv, tbf);
  k_dw3<<<49152, 256, 0, stream>>>(qkv, qdw, qkv2);
  k_attn_part<<<B_ * HEADS * 8, 256, 0, stream>>>(qkv2, attp);
  k_attn_fin<<<B_ * HEADS, 256, 0, stream>>>(attp, temp, attw);
  k_attn_apply<<<B_ * HEADS * 64, 256, 0, stream>>>(qkv2, attw, aob);
  k_dynconv<<<B_ * GRP * 256, 256, 0, stream>>>(aob, tbf, w2, b2, dyn);
  k_idwt_proj<<<512 * B_, 256, 0, stream>>>(dyn, pw, out);
}
